// LlamaAttention_60954175865192
// MI455X (gfx1250) — matrix operand order repaired and checked by reading, not compiled, not run on hardware
//
#include <hip/hip_runtime.h>

// ---------------------------------------------------------------------------
// LLaMA single-token decode attention for MI455X (gfx1250, wave32, WMMA).
//
// Memory-bound workload (~430 MB streamed vs ~3.8 GFLOP): fp32 weights / KV
// are converted to bf16 in registers after load (zero extra bandwidth) and
// all matrix math runs on v_wmma_f32_16x16x32_bf16 with f32 accumulation.
// This revision: native bf16 converts, b128 fragment loads, compile-time
// strides so global loads use immediate offsets (no serialized addr chains).
// ---------------------------------------------------------------------------

#define B_    32
#define HID_  4096
#define NH_   32
#define NKV_  8
#define D_    128
#define GQ_   4        // NH/NKV
#define BS_   64
#define BPS_  32
#define MAXKV_ 2048

typedef __attribute__((ext_vector_type(16))) __bf16  v16bf;
typedef __attribute__((ext_vector_type(8)))  float   v8f;
typedef __attribute__((ext_vector_type(2)))  __bf16  bf16x2;
typedef __attribute__((ext_vector_type(4)))  float   f32x4;
typedef __attribute__((ext_vector_type(4)))  unsigned int u32x4;

union FragU {
    v16bf v;
    unsigned int u[8];
    u32x4 q[2];
};

// Native f32 -> bf16 (clang lowers to the gfx1250 cvt instructions)
__device__ __forceinline__ unsigned int pack_bf16(float lo, float hi) {
    union { bf16x2 h; unsigned int u; } r;
    r.h[0] = (__bf16)lo;
    r.h[1] = (__bf16)hi;
    return r.u;
}

__device__ __forceinline__ unsigned short f2bf(float f) {
    union { __bf16 h; unsigned short s; } r;
    r.h = (__bf16)f;
    return r.s;
}

// u[v] = pack(p[2v], p[2v+1]) : 16 consecutive floats -> B-fragment half-row
__device__ __forceinline__ void pack8_contig(FragU& f, const float* __restrict__ p) {
#pragma unroll
    for (int j = 0; j < 4; ++j) {
        f32x4 t = *(const f32x4*)(p + 4 * j);
        f.u[2 * j]     = pack_bf16(t.x, t.y);
        f.u[2 * j + 1] = pack_bf16(t.z, t.w);
    }
}

// A-fragment layout: u[0..3] <- floats p[0..7], u[4..7] <- floats p[16..23]
__device__ __forceinline__ void pack8_agap(FragU& f, const float* __restrict__ p) {
#pragma unroll
    for (int j = 0; j < 2; ++j) {
        f32x4 t0 = *(const f32x4*)(p + 16 * j);
        f32x4 t1 = *(const f32x4*)(p + 16 * j + 4);
        f.u[4 * j]     = pack_bf16(t0.x, t0.y);
        f.u[4 * j + 1] = pack_bf16(t0.z, t0.w);
        f.u[4 * j + 2] = pack_bf16(t1.x, t1.y);
        f.u[4 * j + 3] = pack_bf16(t1.z, t1.w);
    }
}

// ---------------------------------------------------------------------------
// fp32 -> bf16 convert
// ---------------------------------------------------------------------------
__global__ void cvt_bf16_kernel(const float* __restrict__ in,
                                unsigned short* __restrict__ out, int n) {
    int i = blockIdx.x * blockDim.x + threadIdx.x;
    if (i < n) out[i] = f2bf(in[i]);
}

// ---------------------------------------------------------------------------
// GEMM: out[32][N] = A_bf16[32][K] @ W_f32[K][N]
// One wave per 16-wide N tile; two stacked 16x16 C tiles (M=32) share the
// B fragment so each weight element is loaded exactly once. K,N are template
// parameters so all strided loads use immediate offsets off one base pointer.
// ---------------------------------------------------------------------------
template <int K, int N>
__global__ __launch_bounds__(256)
void gemm_bf16_kernel(const unsigned short* __restrict__ A,
                      const float* __restrict__ W,
                      float* __restrict__ out) {
    int wave = blockIdx.x * (blockDim.x >> 5) + (threadIdx.x >> 5);
    int lane = threadIdx.x & 31;
    int n0 = wave * 16;
    if (n0 >= N) return;

    int m  = lane & 15;
    int kh = lane >> 4;            // lane half: 0 or 1
    int n  = n0 + (lane & 15);

    v8f c0 = {}; v8f c1 = {};

    // A fragment, contiguous k order (B uses the same): lane m, VGPR v holds K pair at
    // kh*16 + v*2 -> 16 contiguous K per lane half.
    const unsigned short* a0p = A + (size_t)m * K + (kh << 4);
    const unsigned short* a1p = a0p + (size_t)16 * K;
    // B fragment, same contiguous k order as A: lanes 0-15 hold
    // K=0..15, lanes 16-31 hold K=16..31; VGPR v -> rows 2v, 2v+1.
    const float* wp = W + (size_t)(kh << 4) * N + n;

    for (int k0 = 0; k0 < K; k0 += 32) {
        FragU a0, a1, b;
        a0.q[0] = *(const u32x4*)(a0p);
        a0.q[1] = *(const u32x4*)(a0p + 8);
        a1.q[0] = *(const u32x4*)(a1p);
        a1.q[1] = *(const u32x4*)(a1p + 8);
#pragma unroll
        for (int v = 0; v < 8; ++v)
            b.u[v] = pack_bf16(wp[(size_t)(2 * v) * N], wp[(size_t)(2 * v + 1) * N]);

        if (k0 + 32 < K)   // speculative prefetch of next weight tile
            __builtin_prefetch(wp + (size_t)32 * N, 0, 1);

        c0 = __builtin_amdgcn_wmma_f32_16x16x32_bf16(false, a0.v, false, b.v,
                                                     (short)0, c0, false, false);
        c1 = __builtin_amdgcn_wmma_f32_16x16x32_bf16(false, a1.v, false, b.v,
                                                     (short)0, c1, false, false);
        a0p += 32;
        a1p += 32;
        wp  += (size_t)32 * N;
    }

    int mb = kh * 8;   // C layout: VGPR r, lanes 0-15 -> M=r, lanes 16-31 -> M=r+8
#pragma unroll
    for (int r = 0; r < 8; ++r) {
        out[(size_t)(mb + r) * N + n]      = c0[r];
        out[(size_t)(mb + r + 16) * N + n] = c1[r];
    }
}

// ---------------------------------------------------------------------------
// RoPE at position kv_lens[b]; 1/sqrt(D) folded into q; q emitted bf16,
// k_new emitted fp32 (stands in for a cache row in the attention kernel).
// ---------------------------------------------------------------------------
__global__ void rope_kernel(const float* __restrict__ qf,
                            const float* __restrict__ kf,
                            const int* __restrict__ kv_lens,
                            unsigned short* __restrict__ q_bf,
                            float* __restrict__ k_rope) {
    const int QN = B_ * NH_ * 64;     // 65536 rotation pairs for q
    const int KN = B_ * NKV_ * 64;    // 16384 for k
    int idx = blockIdx.x * blockDim.x + threadIdx.x;
    if (idx >= QN + KN) return;

    const float kLog1e4_64 = 9.210340371976184f / 64.0f;
    const float scale = 0.08838834764831845f;   // 1/sqrt(128)

    if (idx < QN) {
        int d = idx & 63, h = (idx >> 6) & 31, b = idx >> 11;
        float pos = (float)kv_lens[b];
        float ang = pos * __expf(-(float)d * kLog1e4_64);
        float c = __cosf(ang), s = __sinf(ang);
        const float* src = qf + ((size_t)b * NH_ + h) * D_;
        unsigned short* dst = q_bf + ((size_t)b * NH_ + h) * D_;
        float x1 = src[d], x2 = src[d + 64];
        dst[d]      = f2bf((x1 * c - x2 * s) * scale);
        dst[d + 64] = f2bf((x2 * c + x1 * s) * scale);
    } else {
        int j = idx - QN;
        int d = j & 63, h = (j >> 6) & 7, b = j >> 9;
        float pos = (float)kv_lens[b];
        float ang = pos * __expf(-(float)d * kLog1e4_64);
        float c = __cosf(ang), s = __sinf(ang);
        const float* src = kf + ((size_t)b * NKV_ + h) * D_;
        float* dst = k_rope + ((size_t)b * NKV_ + h) * D_;
        float x1 = src[d], x2 = src[d + 64];
        dst[d]      = x1 * c - x2 * s;
        dst[d + 64] = x2 * c + x1 * s;
    }
}

// ---------------------------------------------------------------------------
// Flash-decode paged GQA attention. One workgroup per (b, kv_head); the 8
// waves split the token range in 32-token chunks (round robin), each keeping
// an online-softmax partial; merged through LDS at the end.
// ---------------------------------------------------------------------------
__device__ __forceinline__ const float* kv_row(const float* __restrict__ cache,
                                               const float* __restrict__ newrow,
                                               const int* __restrict__ block_offsets,
                                               int b, int kvh, int t, int kvlen) {
    if (t == kvlen) return newrow + ((size_t)b * NKV_ + kvh) * D_;
    int blk = block_offsets[b * BPS_ + (t >> 6)];
    return cache + (((size_t)blk * BS_ + (t & 63)) * NKV_ + kvh) * D_;
}

__global__ __launch_bounds__(256)
void attn_kernel(const float* __restrict__ k_cache,
                 const float* __restrict__ v_cache,
                 const int* __restrict__ block_offsets,
                 const int* __restrict__ kv_lens,
                 const unsigned short* __restrict__ q_bf,   // [B][NH][D], pre-scaled
                 const float* __restrict__ k_new,           // [B][NKV][D]
                 const float* __restrict__ v_new,           // [B][NKV][D]
                 unsigned short* __restrict__ attn_bf) {    // [B][NH][D]
    int b    = blockIdx.x >> 3;
    int kvh  = blockIdx.x & 7;
    int wave = threadIdx.x >> 5;
    int lane = threadIdx.x & 31;
    int kvlen = kv_lens[b];
    int S = kvlen + 1;            // valid tokens: 0..kvlen inclusive

    __shared__ float lds_acc[8][GQ_][D_];
    __shared__ float lds_m[8][GQ_];
    __shared__ float lds_l[8][GQ_];
    __shared__ float lds_p[8][16][32];   // P staging: [rows][tokens]

    int m  = lane & 15;
    int kh = lane >> 4;

    // Preload q A-fragments (4 d-chunks of 32); rows >= GQ_ are zero padding.
    // Clamp the row index (m & 3) so padded lanes never read out of bounds.
    FragU qa[4];
    {
        const unsigned short* qrow =
            q_bf + ((size_t)b * NH_ + kvh * GQ_ + (m & 3)) * D_ + (kh << 4);
#pragma unroll
        for (int dc = 0; dc < 4; ++dc) {
            qa[dc].q[0] = *(const u32x4*)(qrow + dc * 32);
            qa[dc].q[1] = *(const u32x4*)(qrow + dc * 32 + 8);
            if (m >= GQ_) { qa[dc].q[0] = (u32x4)0u; qa[dc].q[1] = (u32x4)0u; }
        }
    }

    float mrow[GQ_], lrow[GQ_];
#pragma unroll
    for (int r = 0; r < GQ_; ++r) { mrow[r] = -__builtin_inff(); lrow[r] = 0.f; }
    v8f acc[8] = {};   // out[4..16 pad][128], C layout, 8 d-tiles

    int nchunks = (S + 31) >> 5;
    for (int ch = wave; ch < nchunks; ch += 8) {
        int s0 = ch << 5;

        // ---- scores: q (A) x K^T (B, tokens in N) -----------------------
        v8f c0 = {}, c1 = {};
        int t0 = s0 + (lane & 15);
        int t1 = t0 + 16;
        // Per-lane K row base (+kh*16 floats): each d-chunk is then 16
        // consecutive floats -> four b128 loads with immediate offsets.
        const float* kr0 = kv_row(k_cache, k_new, block_offsets, b, kvh, t0, kvlen) + (kh << 4);
        const float* kr1 = kv_row(k_cache, k_new, block_offsets, b, kvh, t1, kvlen) + (kh << 4);
#pragma unroll
        for (int dc = 0; dc < 4; ++dc) {
            FragU kb0, kb1;
            pack8_contig(kb0, kr0 + dc * 32);
            pack8_contig(kb1, kr1 + dc * 32);
            c0 = __builtin_amdgcn_wmma_f32_16x16x32_bf16(false, qa[dc].v, false, kb0.v,
                                                         (short)0, c0, false, false);
            c1 = __builtin_amdgcn_wmma_f32_16x16x32_bf16(false, qa[dc].v, false, kb1.v,
                                                         (short)0, c1, false, false);
        }

        // mask invalid tokens (whole lane = whole token column)
        if (t0 >= S) {
#pragma unroll
            for (int r = 0; r < 8; ++r) c0[r] = -1e30f;
        }
        if (t1 >= S) {
#pragma unroll
            for (int r = 0; r < 8; ++r) c1[r] = -1e30f;
        }

        // ---- online softmax (rows 0..3; 16-lane shuffle reductions) -----
        float alpha[GQ_];
#pragma unroll
        for (int r = 0; r < GQ_; ++r) {
            float mx = fmaxf(c0[r], c1[r]);
#pragma unroll
            for (int off = 8; off >= 1; off >>= 1)
                mx = fmaxf(mx, __shfl_xor(mx, off, 32));
            float mn = fmaxf(mrow[r], mx);
            alpha[r] = __expf(mrow[r] - mn);
            mrow[r] = mn;
        }

        // exp + stage P into LDS (C layout -> [row][token]); padding rows
        // feed only unread output slots, so their values are harmless.
#pragma unroll
        for (int r = 0; r < 8; ++r) {
            float p0 = (r < GQ_) ? __expf(c0[r] - mrow[r]) : 0.f;
            float p1 = (r < GQ_) ? __expf(c1[r] - mrow[r]) : 0.f;
            lds_p[wave][r + kh * 8][lane & 15]        = p0;
            lds_p[wave][r + kh * 8][16 + (lane & 15)] = p1;
            if (r < GQ_) {
                float ps = p0 + p1;
#pragma unroll
                for (int off = 8; off >= 1; off >>= 1)
                    ps += __shfl_xor(ps, off, 32);
                lrow[r] = lrow[r] * alpha[r] + ps;
            }
        }

        // rescale accumulators (rows beyond GQ_ are zero)
#pragma unroll
        for (int dt = 0; dt < 8; ++dt)
#pragma unroll
            for (int r = 0; r < GQ_; ++r) acc[dt][r] *= alpha[r];

        // ---- read P back as A fragment (per-wave LDS, in-order) ---------
        FragU pa;
        pack8_contig(pa, &lds_p[wave][m][0] + (kh << 4));

        // ---- PV: P (A) x V (B, d in N) ----------------------------------
        const float* vp[16];
#pragma unroll
        for (int i = 0; i < 16; ++i)
            vp[i] = kv_row(v_cache, v_new, block_offsets, b, kvh,
                           s0 + kh * 16 + i, kvlen) + (lane & 15);
#pragma unroll
        for (int dt = 0; dt < 8; ++dt) {
            FragU vb;
#pragma unroll
            for (int v = 0; v < 8; ++v)
                vb.u[v] = pack_bf16(vp[2 * v][dt * 16], vp[2 * v + 1][dt * 16]);
            acc[dt] = __builtin_amdgcn_wmma_f32_16x16x32_bf16(false, pa.v, false, vb.v,
                                                              (short)0, acc[dt], false, false);
        }
    }

    // ---- publish per-wave partials --------------------------------------
    if (lane < 16) {
#pragma unroll
        for (int dt = 0; dt < 8; ++dt)
#pragma unroll
            for (int r = 0; r < GQ_; ++r)
                lds_acc[wave][r][dt * 16 + lane] = acc[dt][r];
        if (lane == 0) {
#pragma unroll
            for (int r = 0; r < GQ_; ++r) {
                lds_m[wave][r] = mrow[r];
                lds_l[wave][r] = lrow[r];
            }
        }
    }
    __syncthreads();

    // ---- flash-decode merge across the 8 waves --------------------------
    for (int p = threadIdx.x; p < GQ_ * D_; p += 256) {
        int g = p >> 7, d = p & 127;
        float mg = -__builtin_inff();
#pragma unroll
        for (int w = 0; w < 8; ++w) mg = fmaxf(mg, lds_m[w][g]);
        float ls = 0.f, vs = 0.f;
#pragma unroll
        for (int w = 0; w < 8; ++w) {
            float sc = __expf(lds_m[w][g] - mg);
            ls += sc * lds_l[w][g];
            vs += sc * lds_acc[w][g][d];
        }
        attn_bf[((size_t)b * NH_ + kvh * GQ_ + g) * D_ + d] = f2bf(vs / ls);
    }
}

// ---------------------------------------------------------------------------
// launch
// ---------------------------------------------------------------------------
extern "C" void kernel_launch(void* const* d_in, const int* in_sizes, int n_in,
                              void* d_out, int out_size, void* d_ws, size_t ws_size,
                              hipStream_t stream) {
    (void)in_sizes; (void)n_in; (void)out_size; (void)ws_size;
    const float* hs        = (const float*)d_in[0];
    const float* k_cache   = (const float*)d_in[1];
    const float* v_cache   = (const float*)d_in[2];
    const int*   block_off = (const int*)d_in[3];
    const int*   kv_lens   = (const int*)d_in[4];
    const float* wq        = (const float*)d_in[5];
    const float* wk        = (const float*)d_in[6];
    const float* wv        = (const float*)d_in[7];
    const float* wo        = (const float*)d_in[8];
    float* out = (float*)d_out;

    char* ws = (char*)d_ws;
    size_t off = 0;
    auto alloc = [&](size_t bytes) -> void* {
        void* p = ws + off;
        off += (bytes + 255) & ~(size_t)255;
        return p;
    };
    unsigned short* hs_bf   = (unsigned short*)alloc((size_t)B_ * HID_ * 2);
    float*          q_f     = (float*)alloc((size_t)B_ * NH_ * D_ * 4);
    float*          k_f     = (float*)alloc((size_t)B_ * NKV_ * D_ * 4);
    float*          v_f     = (float*)alloc((size_t)B_ * NKV_ * D_ * 4);
    unsigned short* q_bf    = (unsigned short*)alloc((size_t)B_ * NH_ * D_ * 2);
    float*          k_rope  = (float*)alloc((size_t)B_ * NKV_ * D_ * 4);
    unsigned short* attn_bf = (unsigned short*)alloc((size_t)B_ * NH_ * D_ * 2);

    cvt_bf16_kernel<<<(B_ * HID_) / 256, 256, 0, stream>>>(hs, hs_bf, B_ * HID_);

    // QKV projections: one 16-wide N tile per wave, 8 waves per block
    gemm_bf16_kernel<HID_, NH_ * D_><<<(NH_ * D_ / 16) / 8, 256, 0, stream>>>(hs_bf, wq, q_f);
    gemm_bf16_kernel<HID_, NKV_ * D_><<<(NKV_ * D_ / 16) / 8, 256, 0, stream>>>(hs_bf, wk, k_f);
    gemm_bf16_kernel<HID_, NKV_ * D_><<<(NKV_ * D_ / 16) / 8, 256, 0, stream>>>(hs_bf, wv, v_f);

    rope_kernel<<<(B_ * NH_ * 64 + B_ * NKV_ * 64) / 256, 256, 0, stream>>>(
        q_f, k_f, kv_lens, q_bf, k_rope);

    attn_kernel<<<B_ * NKV_, 256, 0, stream>>>(k_cache, v_cache, block_off, kv_lens,
                                               q_bf, k_rope, v_f, attn_bf);

    gemm_bf16_kernel<NH_ * D_, HID_><<<(HID_ / 16) / 8, 256, 0, stream>>>(attn_bf, wo, out);
}
